// SharedDynamicEdgeConv_16140487098438
// MI455X (gfx1250) — matrix operand order repaired and checked by reading, not compiled, not run on hardware
//
#include <hip/hip_runtime.h>
#include <hip/hip_bf16.h>

// ---------------------------------------------------------------------------
// SharedDynamicEdgeConv for MI455X (gfx1250, wave32, WMMA + async-to-LDS)
//   x(8192x3) -> EdgeConv(k=16, 6->128->128) -> EdgeConv(k=8, 256->256->256)
// Heavy GEMMs (knn2 Gram matrix, layer2 MLP) use v_wmma_f32_16x16x32_f16.
// knn2 stages B tiles with GLOBAL_LOAD_ASYNC_TO_LDS_B128 (double buffered).
// ---------------------------------------------------------------------------

#define NPTS 8192
#define INC  3
#define HID  128
#define OUTC 256
#define K1   16
#define K2   8

typedef __attribute__((ext_vector_type(16))) _Float16 v16h;
typedef __attribute__((ext_vector_type(8)))  _Float16 v8h;
typedef __attribute__((ext_vector_type(8)))  float    v8f;

template <int K>
__device__ __forceinline__ void topk_insert(float (&best)[K], int (&bidx)[K],
                                            float d, int j) {
    // best[] kept sorted ascending; caller guarantees d < best[K-1].
    float cv = d; int ci = j;
#pragma unroll
    for (int t = 0; t < K; ++t) {
        if (cv < best[t]) {
            float tv = best[t]; int ti = bidx[t];
            best[t] = cv; bidx[t] = ci;
            cv = tv; ci = ti;
        }
    }
}

// Low 32 bits of a flat pointer into __shared__ give the LDS byte offset
// (flat->LDS mapping truncates the upper bits; see CDNA5 aperture rules).
__device__ __forceinline__ unsigned lds_off(const void* p) {
    return (unsigned)(uintptr_t)p;
}

// ---------------------------------------------------------------------------
// Kernel 1: exact f32 KNN on 3-dim input, k=16. One thread per point.
// ---------------------------------------------------------------------------
__global__ void knn1_kernel(const float* __restrict__ x, int* __restrict__ idx1) {
    int i = blockIdx.x * blockDim.x + threadIdx.x;
    float xi0 = x[i * 3 + 0], xi1 = x[i * 3 + 1], xi2 = x[i * 3 + 2];
    float best[K1]; int bidx[K1];
#pragma unroll
    for (int t = 0; t < K1; ++t) { best[t] = 3.0e38f; bidx[t] = 0; }
    for (int j = 0; j < NPTS; ++j) {
        if (j == i) continue;                       // reference sets diag = inf
        float d0 = x[j * 3 + 0] - xi0;
        float d1 = x[j * 3 + 1] - xi1;
        float d2 = x[j * 3 + 2] - xi2;
        float d = d0 * d0 + d1 * d1 + d2 * d2;
        if (d < best[K1 - 1]) topk_insert<K1>(best, bidx, d, j);
    }
#pragma unroll
    for (int t = 0; t < K1; ++t) idx1[i * K1 + t] = bidx[t];
}

// ---------------------------------------------------------------------------
// Kernel 2: layer-1 edge MLP, f32 VALU (only ~4.5 GFLOP).
//   e = [xi, xj-xi] (6) -> relu(e@W1a+b1a) (128) -> @W1b+b1b (128) -> max_k
// blockDim = 128 (one thread per channel). Weights cached in LDS.
// ---------------------------------------------------------------------------
__global__ void layer1_kernel(const float* __restrict__ x,
                              const int* __restrict__ idx1,
                              const float* __restrict__ W1a,
                              const float* __restrict__ b1a,
                              const float* __restrict__ W1b,
                              const float* __restrict__ b1b,
                              float* __restrict__ h,
                              _Float16* __restrict__ hh) {
    __shared__ float     sW1a[6 * HID];
    __shared__ float     sb1a[HID];
    __shared__ float     sb1b[HID];
    __shared__ _Float16  sW1b[HID * HID];   // 32 KB, f16 to keep LDS modest
    __shared__ float     se[K1 * 8];        // 6 features padded to 8
    __shared__ float     sh1[K1 * HID];     // 8 KB

    int t = threadIdx.x;                     // 0..127 == channel
    for (int i = t; i < 6 * HID; i += HID) sW1a[i] = W1a[i];
    sb1a[t] = b1a[t];
    sb1b[t] = b1b[t];
    for (int i = t; i < HID * HID; i += HID) sW1b[i] = (_Float16)W1b[i];
    __syncthreads();

    for (int i = blockIdx.x; i < NPTS; i += gridDim.x) {
        if (t < K1) {
            int j = idx1[i * K1 + t];
            float a0 = x[i * 3 + 0], a1 = x[i * 3 + 1], a2 = x[i * 3 + 2];
            se[t * 8 + 0] = a0;
            se[t * 8 + 1] = a1;
            se[t * 8 + 2] = a2;
            se[t * 8 + 3] = x[j * 3 + 0] - a0;
            se[t * 8 + 4] = x[j * 3 + 1] - a1;
            se[t * 8 + 5] = x[j * 3 + 2] - a2;
        }
        __syncthreads();
        for (int kk = 0; kk < K1; ++kk) {
            float acc = sb1a[t];
#pragma unroll
            for (int d = 0; d < 6; ++d) acc += se[kk * 8 + d] * sW1a[d * HID + t];
            sh1[kk * HID + t] = fmaxf(acc, 0.0f);
        }
        __syncthreads();
        float m = -3.0e38f;
        for (int kk = 0; kk < K1; ++kk) {
            float acc = sb1b[t];
            for (int d = 0; d < HID; ++d)
                acc += sh1[kk * HID + d] * (float)sW1b[d * HID + t];
            m = fmaxf(m, acc);
        }
        h[(size_t)i * HID + t]  = m;
        hh[(size_t)i * HID + t] = (_Float16)m;
        __syncthreads();
    }
}

// ---------------------------------------------------------------------------
// Kernel 3: squared norms of h rows (for knn2 ranking).
// ---------------------------------------------------------------------------
__global__ void sqnorm_kernel(const float* __restrict__ h, float* __restrict__ sqn) {
    int i = blockIdx.x * blockDim.x + threadIdx.x;
    if (i < NPTS) {
        float s = 0.0f;
        for (int c = 0; c < HID; ++c) {
            float v = h[(size_t)i * HID + c];
            s += v * v;
        }
        sqn[i] = s;
    }
}

// ---------------------------------------------------------------------------
// Kernel 4: convert layer-2 weights to f16, N-major (B-fragment friendly):
//   W2aT[n*256 + k] = W2a[k*256 + n]
// ---------------------------------------------------------------------------
__global__ void convw_kernel(const float* __restrict__ W2a,
                             const float* __restrict__ W2b,
                             _Float16* __restrict__ W2aT,
                             _Float16* __restrict__ W2bT) {
    int i = blockIdx.x * blockDim.x + threadIdx.x;   // 65536 threads
    int k = i >> 8, n = i & 255;
    W2aT[n * 256 + k] = (_Float16)W2a[k * 256 + n];
    W2bT[n * 256 + k] = (_Float16)W2b[k * 256 + n];
}

// ---------------------------------------------------------------------------
// Kernel 5: fused Gram-matrix + top-8 KNN on h (128-dim) via WMMA f16.
// blockDim=256 (8 waves). Each wave owns a 16-row strip; A fragments stay in
// registers for the whole 8192-column sweep. The shared 16x128 B tile is
// staged once per block into double-buffered LDS by wave 0 using
// GLOBAL_LOAD_ASYNC_TO_LDS_B128 (overlapped with compute on the other
// buffer), instead of each of the 8 waves pulling it from L2.
// Ranking key: sqn[j] - 2*h_i.h_j (the +sqn[i] term is row-constant).
// ---------------------------------------------------------------------------
__global__ void knn2_kernel(const _Float16* __restrict__ hh,
                            const float* __restrict__ sqn,
                            int* __restrict__ idx2) {
    __shared__ __align__(16) _Float16 sb[2][16][HID];   // 2 x 4 KB B staging
    __shared__ __align__(16) float    sc[8][16][16];    // per-wave score tile

    int lane = threadIdx.x & 31;
    int wave = threadIdx.x >> 5;
    int l16  = lane & 15;
    int hiH  = lane >> 4;                            // 0 or 1
    int row0 = blockIdx.x * 128 + wave * 16;
    int myrow = row0 + l16;                          // row owned by lanes 0..15

    // Async-staging lane mapping: each issue q covers 2 rows (512 B);
    // lane -> row parity (lane>>4), 16-byte column chunk (lane&15).
    unsigned st_col  = (unsigned)(l16) * 16u;        // byte offset within row
    unsigned st_rpar = (unsigned)hiH;

    // A fragments (16x128 strip, 4 k-steps of K=32).
    // lane<16 : row M=l16, K = {32ks+0..7, 32ks+16..23}
    // lane>=16: row M=l16, K = {32ks+8..15, 32ks+24..31}
    const _Float16* arow = hh + (size_t)myrow * HID;
    v16h a[4];
#pragma unroll
    for (int ks = 0; ks < 4; ++ks) {
        int kb = 32 * ks + hiH * 8;
        union { v8h half[2]; v16h v; } ua;
        ua.half[0] = *(const v8h*)(arow + kb);
        ua.half[1] = *(const v8h*)(arow + kb + 16);
        a[ks] = ua.v;
    }

    float best[K2]; int bidx[K2];
#pragma unroll
    for (int t = 0; t < K2; ++t) { best[t] = 3.0e38f; bidx[t] = 0; }

    // Prologue: stage tile jt=0 into buffer 0.
    if (wave == 0) {
#pragma unroll
        for (int q = 0; q < 8; ++q) {
            unsigned row  = (unsigned)q * 2u + st_rpar;
            unsigned voff = row * (HID * 2u) + st_col;               // jt = 0
            unsigned loff = lds_off(&sb[0][0][0]) + row * (HID * 2u) + st_col;
            asm volatile("global_load_async_to_lds_b128 %0, %1, %2"
                         :: "v"(loff), "v"(voff), "s"(hh) : "memory");
        }
        asm volatile("s_wait_asynccnt 0" ::: "memory");
    }
    __syncthreads();

    for (int jt = 0; jt < NPTS; jt += 16) {
        int buf = (jt >> 4) & 1;
        // Kick off the async copy of the NEXT tile into the other buffer.
        if (wave == 0 && (jt + 16) < NPTS) {
            unsigned gbase = (unsigned)(jt + 16) * (HID * 2u);
            unsigned lbase = lds_off(&sb[buf ^ 1][0][0]);
#pragma unroll
            for (int q = 0; q < 8; ++q) {
                unsigned row  = (unsigned)q * 2u + st_rpar;
                unsigned voff = gbase + row * (HID * 2u) + st_col;
                unsigned loff = lbase + row * (HID * 2u) + st_col;
                asm volatile("global_load_async_to_lds_b128 %0, %1, %2"
                             :: "v"(loff), "v"(voff), "s"(hh) : "memory");
            }
        }

        // 16x16 dot tile from the CURRENT staged buffer.
        v8f c = {};
#pragma unroll
        for (int ks = 0; ks < 4; ++ks) {
            // B fragment: column N=l16 (point jt+l16), K = {32ks+hiH*8+0..7, 32ks+16+hiH*8+0..7}
            const _Float16* bp = &sb[buf][l16][32 * ks + hiH * 8];
            union { v8h half[2]; v16h v; } ub;
            ub.half[0] = *(const v8h*)(bp);
            ub.half[1] = *(const v8h*)(bp + 16);
            c = __builtin_amdgcn_wmma_f32_16x16x32_f16(
                    false, a[ks], false, ub.v, (short)0, c, false, false);
        }
        // C layout: VGPR r, lanes 0..15 -> (M=r, N=lane); lanes 16..31 -> (M=r+8)
        float sj = sqn[jt + l16];
#pragma unroll
        for (int r = 0; r < 8; ++r) {
            int m = r + hiH * 8;
            sc[wave][m][l16] = sj - 2.0f * c[r];
        }
        __syncthreads();
        if (lane < 16) {
#pragma unroll
            for (int n = 0; n < 16; ++n) {
                int j = jt + n;
                if (j == myrow) continue;
                float s = sc[wave][l16][n];
                if (s < best[K2 - 1]) topk_insert<K2>(best, bidx, s, j);
            }
        }
        // Publish the next buffer: issuing wave drains ASYNCcnt, then all sync.
        if (wave == 0) {
            asm volatile("s_wait_asynccnt 0" ::: "memory");
        }
        __syncthreads();
    }
    if (lane < 16) {
#pragma unroll
        for (int t = 0; t < K2; ++t) idx2[(size_t)myrow * K2 + t] = bidx[t];
    }
}

// ---------------------------------------------------------------------------
// Kernel 6: layer-2 edge MLP via WMMA. One wave handles 2 points x 8 edges
// = an M=16 tile. e=[h_i, h_j-h_i] (K=256) built directly in A-fragment
// registers; GEMM1 -> relu -> LDS (f16) -> GEMM2 -> per-lane 8-row max.
// blockDim=128 (4 waves => 8 points/block, 32 KB LDS).
// ---------------------------------------------------------------------------
__global__ void layer2_kernel(const _Float16* __restrict__ hh,
                              const int* __restrict__ idx2,
                              const _Float16* __restrict__ W2aT,
                              const float* __restrict__ b2a,
                              const _Float16* __restrict__ W2bT,
                              const float* __restrict__ b2b,
                              float* __restrict__ out) {
    __shared__ __align__(16) _Float16 sh1[4][16][OUTC];   // 4 x 8 KB

    int lane = threadIdx.x & 31;
    int wave = threadIdx.x >> 5;
    int l16  = lane & 15;
    int hiH  = lane >> 4;
    int p0   = blockIdx.x * 8 + wave * 2;                 // 2 points per wave

    int m = l16;                                          // tile row 0..15
    int p = p0 + (m >> 3);                                // rows 0-7 -> p0, 8-15 -> p1
    int e = m & 7;
    int j = idx2[(size_t)p * K2 + e];
    const _Float16* hp = hh + (size_t)p * HID;
    const _Float16* hj = hh + (size_t)j * HID;

    // Edge-feature A fragments: K 0..127 = h_i ; K 128..255 = h_j - h_i
    v16h a[8];
#pragma unroll
    for (int ks = 0; ks < 8; ++ks) {
        int kb = 32 * ks + hiH * 8;                       // first 8-half chunk
        union { v8h half[2]; v16h v; } ua;
        if (ks < 4) {
            ua.half[0] = *(const v8h*)(hp + kb);
            ua.half[1] = *(const v8h*)(hp + kb + 16);
        } else {
            int kb2 = kb - 128;
            ua.half[0] = *(const v8h*)(hj + kb2)      - *(const v8h*)(hp + kb2);
            ua.half[1] = *(const v8h*)(hj + kb2 + 16) - *(const v8h*)(hp + kb2 + 16);
        }
        a[ks] = ua.v;
    }

    // GEMM1: 16x256 = relu(E @ W2a + b2a), tile over N, accumulate over K.
    for (int nt = 0; nt < 16; ++nt) {
        float bias = b2a[nt * 16 + l16];
        v8f c;
#pragma unroll
        for (int r = 0; r < 8; ++r) c[r] = bias;
#pragma unroll
        for (int ks = 0; ks < 8; ++ks) {
            const _Float16* bp = W2aT + (size_t)(nt * 16 + l16) * OUTC
                                 + 32 * ks + hiH * 8;
            union { v8h half[2]; v16h v; } ub;
            ub.half[0] = *(const v8h*)(bp);
            ub.half[1] = *(const v8h*)(bp + 16);
            c = __builtin_amdgcn_wmma_f32_16x16x32_f16(
                    false, a[ks], false, ub.v, (short)0, c, false, false);
        }
#pragma unroll
        for (int r = 0; r < 8; ++r) {
            int mm = r + hiH * 8;
            sh1[wave][mm][nt * 16 + l16] = (_Float16)fmaxf(c[r], 0.0f);
        }
    }
    __syncthreads();

    // Re-fragment h1 from LDS as A for GEMM2.
    v16h a2[8];
#pragma unroll
    for (int ks = 0; ks < 8; ++ks) {
        int kb = 32 * ks + hiH * 8;
        const _Float16* rp = &sh1[wave][l16][0];
        union { v8h half[2]; v16h v; } ua;
        ua.half[0] = *(const v8h*)(rp + kb);
        ua.half[1] = *(const v8h*)(rp + kb + 16);
        a2[ks] = ua.v;
    }

    // GEMM2 + k-max: lanes 0..15 hold rows 0..7 (p0), lanes 16..31 rows 8..15 (p1);
    // the 8 accumulator VGPRs per lane ARE the 8 edges of one point -> local max.
    int pout = p0 + hiH;
    for (int nt = 0; nt < 16; ++nt) {
        float bias = b2b[nt * 16 + l16];
        v8f c;
#pragma unroll
        for (int r = 0; r < 8; ++r) c[r] = bias;
#pragma unroll
        for (int ks = 0; ks < 8; ++ks) {
            const _Float16* bp = W2bT + (size_t)(nt * 16 + l16) * OUTC
                                 + 32 * ks + hiH * 8;
            union { v8h half[2]; v16h v; } ub;
            ub.half[0] = *(const v8h*)(bp);
            ub.half[1] = *(const v8h*)(bp + 16);
            c = __builtin_amdgcn_wmma_f32_16x16x32_f16(
                    false, a2[ks], false, ub.v, (short)0, c, false, false);
        }
        float mx = c[0];
#pragma unroll
        for (int r = 1; r < 8; ++r) mx = fmaxf(mx, c[r]);
        out[(size_t)pout * OUTC + nt * 16 + l16] = mx;
    }
}

// ---------------------------------------------------------------------------
extern "C" void kernel_launch(void* const* d_in, const int* in_sizes, int n_in,
                              void* d_out, int out_size, void* d_ws, size_t ws_size,
                              hipStream_t stream) {
    const float* x   = (const float*)d_in[0];
    const float* W1a = (const float*)d_in[1];
    const float* b1a = (const float*)d_in[2];
    const float* W1b = (const float*)d_in[3];
    const float* b1b = (const float*)d_in[4];
    const float* W2a = (const float*)d_in[5];
    const float* b2a = (const float*)d_in[6];
    const float* W2b = (const float*)d_in[7];
    const float* b2b = (const float*)d_in[8];
    float* out = (float*)d_out;

    char* ws = (char*)d_ws;
    size_t off = 0;
    int*      idx1 = (int*)(ws + off);      off += (size_t)NPTS * K1 * sizeof(int);
    float*    h    = (float*)(ws + off);    off += (size_t)NPTS * HID * sizeof(float);
    _Float16* hh   = (_Float16*)(ws + off); off += (size_t)NPTS * HID * sizeof(_Float16);
    float*    sqn  = (float*)(ws + off);    off += (size_t)NPTS * sizeof(float);
    int*      idx2 = (int*)(ws + off);      off += (size_t)NPTS * K2 * sizeof(int);
    _Float16* W2aT = (_Float16*)(ws + off); off += (size_t)OUTC * OUTC * sizeof(_Float16);
    _Float16* W2bT = (_Float16*)(ws + off); off += (size_t)OUTC * OUTC * sizeof(_Float16);

    knn1_kernel  <<<NPTS / 256, 256, 0, stream>>>(x, idx1);
    layer1_kernel<<<64, 128, 0, stream>>>(x, idx1, W1a, b1a, W1b, b1b, h, hh);
    sqnorm_kernel<<<NPTS / 256, 256, 0, stream>>>(h, sqn);
    convw_kernel <<<(OUTC * OUTC) / 256, 256, 0, stream>>>(W2a, W2b, W2aT, W2bT);
    knn2_kernel  <<<NPTS / 128, 256, 0, stream>>>(hh, sqn, idx2);
    layer2_kernel<<<NPTS / 8, 128, 0, stream>>>(hh, idx2, W2aT, b2a, W2bT, b2b, out);
}
